// CreditDecomposer_34033320853703
// MI455X (gfx1250) — matrix operand order repaired and checked by reading, not compiled, not run on hardware
//
#include <hip/hip_runtime.h>
#include <hip/hip_bf16.h>

typedef __bf16 bf16;
typedef __attribute__((ext_vector_type(16))) __bf16 v16bf;
typedef __attribute__((ext_vector_type(8)))  float  v8f;

#define DIM   512
#define MT    32          // rows per block
#define CSTR  1040        // comb row stride (bf16 elems): 1024 + 16 pad
#define QSTR  520         // q row stride (f32 elems): 512 + 8 pad
#define HSTR  68          // h row stride (f32 elems): 64 + 4 pad

// workspace layout (bytes)
#define WS_KBUF 0                          // k vectors: 4 x 512 f32
#define WS_VBUF (4*512*4)                  // v vectors: 4 x 512 f32
#define WS_WQ   (2*4*512*4)                // Wq bf16, pre-transposed [ks][n][kk]
#define WS_W1   (WS_WQ + 512*512*2)        // W1 bf16, pre-transposed [ks][n][kk]

// dynamic LDS layout (bytes). No static __shared__ anywhere, so the dynamic
// segment starts at LDS offset 0 and these offsets are raw DS addresses
// usable directly by the async-DMA instructions.
#define L_COMB  0
#define L_Q     (MT*CSTR*2)                         // 66560
#define L_WSL   (L_Q + MT*QSTR*4)                   // 133120 (2x32KB dbl buf)
#define L_H     (L_WSL + 2*512*32*2)                // 198656
#define SMEM_BYTES (L_H + MT*HSTR*4)                // 207360

// Async global->LDS DMA (VGLOBAL encoding, GVS mode: SGPR64 base + VGPR32
// byte offset). Tracked by ASYNCcnt; waited with s_wait_asynccnt.
__device__ __forceinline__ void async_copy_b128(unsigned ldsByteOff,
                                                const void* gbase,
                                                unsigned gByteOff) {
    asm volatile("global_load_async_to_lds_b128 %0, %1, %2"
                 :: "v"(ldsByteOff), "v"(gByteOff), "s"(gbase)
                 : "memory");
}
__device__ __forceinline__ void wait_async0() {
    asm volatile("s_wait_asynccnt 0x0" ::: "memory");
}

// ---------------------------------------------------------------------------
// Prep 1: convert Wq (512x512) and W1 (1024x64) f32 -> bf16, pre-transposed
// into K-slab layout: dst[((ks*N + n)*32) + kk] = src[(ks*32+kk)*N + n]
// so each 32-K slab is a straight contiguous copy (async-DMA friendly) and
// B-fragment lanes read 16 contiguous bf16 (two b128 LDS reads).
// ---------------------------------------------------------------------------
__global__ __launch_bounds__(256) void conv_weights(
    const float* __restrict__ Wq, const float* __restrict__ W1,
    bf16* __restrict__ wq16, bf16* __restrict__ w116)
{
    int i = blockIdx.x * 256 + threadIdx.x;
    if (i < 512*512) {
        int kk = i & 31;
        int n  = (i >> 5) & 511;
        int ks = i >> 14;
        wq16[i] = (bf16)Wq[(ks*32 + kk)*512 + n];
    } else if (i < 512*512 + 1024*64) {
        int j  = i - 512*512;
        int kk = j & 31;
        int n  = (j >> 5) & 63;
        int ks = j >> 11;
        w116[j] = (bf16)W1[(ks*32 + kk)*64 + n];
    }
}

// ---------------------------------------------------------------------------
// Prep 2: k = emb[seq] @ Wk + bk ; v = emb[seq] @ Wv + bv   (4 x 512 each)
// Tiny (4 MFLOP): one block, scalar dots, L2-resident thereafter.
// ---------------------------------------------------------------------------
__global__ __launch_bounds__(512) void prep_kv(
    const int* __restrict__ seq, const float* __restrict__ emb,
    const float* __restrict__ Wk, const float* __restrict__ bk,
    const float* __restrict__ Wv, const float* __restrict__ bv,
    float* __restrict__ kbuf, float* __restrict__ vbuf)
{
    int c = threadIdx.x;   // column 0..511
    #pragma unroll
    for (int p = 0; p < 4; ++p) {
        const float* pr = emb + (size_t)seq[p] * DIM;
        float sk = bk[c], sv = bv[c];
        for (int d = 0; d < DIM; ++d) {
            float e = pr[d];
            sk += e * Wk[d*DIM + c];
            sv += e * Wv[d*DIM + c];
        }
        kbuf[p*DIM + c] = sk;
        vbuf[p*DIM + c] = sv;
    }
}

// ---------------------------------------------------------------------------
// Main fused kernel: 32 rows per block, 256 threads (8 wave32).
// ---------------------------------------------------------------------------
__global__ __launch_bounds__(256) void credit_main(
    const float* __restrict__ x,   const bf16* __restrict__ wq16,
    const float* __restrict__ bq,  const float* __restrict__ kbuf,
    const float* __restrict__ vbuf,const bf16* __restrict__ w116,
    const float* __restrict__ b1,  const float* __restrict__ W2,
    const float* __restrict__ b2,
    float* __restrict__ credit_out, float* __restrict__ attn_out)
{
    extern __shared__ char smem[];
    bf16*  comb = (bf16*) (smem + L_COMB);   // [MT][CSTR] bf16: x | weighted
    float* qls  = (float*)(smem + L_Q);      // [MT][QSTR] f32
    bf16*  wsl  = (bf16*) (smem + L_WSL);    // weight slab, double-buffered
    float* hls  = (float*)(smem + L_H);      // [MT][HSTR] f32

    const int t    = threadIdx.x;
    const int lane = t & 31;
    const int w    = t >> 5;
    const int m    = lane & 15;
    const int hi   = lane >> 4;              // 0: lanes 0-15, 1: lanes 16-31
    const size_t rowBase = (size_t)blockIdx.x * MT;

    // ---- Phase A: x tile -> comb[:, 0:512) as bf16 --------------------------
    {
        int r = t >> 3, part = t & 7;
        const float* src = x + (rowBase + r) * DIM + part * 64;
        bf16* dst = comb + r * CSTR + part * 64;
        #pragma unroll
        for (int j = 0; j < 64; j += 4) {
            float4 vv = *(const float4*)(src + j);
            dst[j+0] = (bf16)vv.x; dst[j+1] = (bf16)vv.y;
            dst[j+2] = (bf16)vv.z; dst[j+3] = (bf16)vv.w;
        }
    }

    // ---- Phase B: q = x @ Wq (+bq), M=32 N=512 K=512 via WMMA bf16 ----------
    // Weight slabs stream via async global->LDS DMA, double-buffered: one
    // barrier per K-step, DMA for slab k+1 overlaps the 8 WMMAs on slab k.
    {
        const int mt = w >> 2;   // 0..1  : 16-row half
        const int ng = w & 3;    // 0..3  : 128-col strip (8 N-tiles)
        v8f acc[8];
        #pragma unroll
        for (int i = 0; i < 8; ++i)
            #pragma unroll
            for (int r = 0; r < 8; ++r) acc[i][r] = 0.0f;

        // prologue: DMA slab 0 into buffer 0 (32KB, 8 x b128 per thread)
        #pragma unroll
        for (int i = 0; i < 8; ++i)
            async_copy_b128((unsigned)(L_WSL + (t + i*256)*16), wq16,
                            (unsigned)((t + i*256)*16));

        for (int ks = 0; ks < 16; ++ks) {
            wait_async0();          // own-wave DMA done
            __syncthreads();        // slab[cur] visible; slab[nxt] free
            const int cur = ks & 1;
            if (ks + 1 < 16) {
                const bf16* gn = wq16 + (size_t)(ks + 1) * 512 * 32;
                #pragma unroll
                for (int i = 0; i < 8; ++i)
                    async_copy_b128(
                        (unsigned)(L_WSL + (1 - cur)*32768 + (t + i*256)*16),
                        gn, (unsigned)((t + i*256)*16));
            }

            // A fragment 16x32: lane half selects K 0..15 vs 16..31
            const bf16* arow = comb + (mt*16 + m) * CSTR + ks*32 + hi*16;
            v16bf a;
            #pragma unroll
            for (int i = 0; i < 8; ++i) { a[i] = arow[i]; a[8+i] = arow[8 + i]; }

            const bf16* wbuf = wsl + cur * 16384;
            // manual unroll keeps reuse_a a literal: hint A-reuse on the
            // first 7 of the 8 WMMAs sharing this A fragment
            #define QWMMA(NT, RA)                                              \
            {                                                                  \
                const bf16* brow = wbuf + (ng*128 + NT*16 + m) * 32 + hi*16;   \
                v16bf b;                                                       \
                _Pragma("unroll")                                              \
                for (int i = 0; i < 16; ++i) b[i] = brow[i];                   \
                acc[NT] = __builtin_amdgcn_wmma_f32_16x16x32_bf16(             \
                    false, a, false, b, (short)0, acc[NT], RA, false);         \
            }
            QWMMA(0, true) QWMMA(1, true) QWMMA(2, true) QWMMA(3, true)
            QWMMA(4, true) QWMMA(5, true) QWMMA(6, true) QWMMA(7, false)
            #undef QWMMA
        }
        // spill q (+bq) to LDS; C layout: M = hi*8 + r, N = m
        #pragma unroll
        for (int nt = 0; nt < 8; ++nt) {
            int col = ng*128 + nt*16 + m;
            float bqv = bq[col];
            #pragma unroll
            for (int r = 0; r < 8; ++r)
                qls[(mt*16 + hi*8 + r) * QSTR + col] = acc[nt][r] + bqv;
        }
    }
    __syncthreads();

    // ---- Phase C: scores (P=4), softmax, weighted-v -> comb[:, 512:) --------
    {
        int r = t >> 3, part = t & 7;     // 8 threads per row, 64 cols each
        const float* qrow = qls + r * QSTR + part * 64;
        const float* k0 = kbuf + 0*DIM + part*64;
        const float* k1 = kbuf + 1*DIM + part*64;
        const float* k2 = kbuf + 2*DIM + part*64;
        const float* k3 = kbuf + 3*DIM + part*64;
        float s0=0.f, s1=0.f, s2=0.f, s3=0.f;
        #pragma unroll 8
        for (int j = 0; j < 64; ++j) {
            float qv = qrow[j];
            s0 += qv*k0[j]; s1 += qv*k1[j]; s2 += qv*k2[j]; s3 += qv*k3[j];
        }
        #pragma unroll
        for (int off = 1; off < 8; off <<= 1) {
            s0 += __shfl_xor(s0, off, 8);
            s1 += __shfl_xor(s1, off, 8);
            s2 += __shfl_xor(s2, off, 8);
            s3 += __shfl_xor(s3, off, 8);
        }
        const float scale = 0.044194173824159216f;  // 1/sqrt(512)
        s0 *= scale; s1 *= scale; s2 *= scale; s3 *= scale;
        float mx = fmaxf(fmaxf(s0, s1), fmaxf(s2, s3));
        float e0 = __expf(s0-mx), e1 = __expf(s1-mx),
              e2 = __expf(s2-mx), e3 = __expf(s3-mx);
        float inv = 1.0f / (e0+e1+e2+e3);
        float a0 = e0*inv, a1 = e1*inv, a2 = e2*inv, a3 = e3*inv;
        if (part == 0) {
            float4 av = make_float4(a0, a1, a2, a3);
            *(float4*)(attn_out + (rowBase + r) * 4) = av;
        }
        const float* v0 = vbuf + 0*DIM + part*64;
        const float* v1 = vbuf + 1*DIM + part*64;
        const float* v2 = vbuf + 2*DIM + part*64;
        const float* v3 = vbuf + 3*DIM + part*64;
        bf16* dst = comb + r * CSTR + DIM + part * 64;
        #pragma unroll 8
        for (int j = 0; j < 64; ++j)
            dst[j] = (bf16)(a0*v0[j] + a1*v1[j] + a2*v2[j] + a3*v3[j]);
    }
    __syncthreads();

    // ---- Phase D: h = relu([x|weighted] @ W1 + b1), M=32 K=1024 N=64 --------
    // Same async double-buffered slab scheme; slab is 4KB (1 b128/thread).
    {
        const int mt = w >> 2;   // 16-row half
        const int nt = w & 3;    // 16-col tile
        v8f acc;
        #pragma unroll
        for (int r = 0; r < 8; ++r) acc[r] = 0.0f;

        async_copy_b128((unsigned)(L_WSL + t*16), w116, (unsigned)(t*16));

        for (int ks = 0; ks < 32; ++ks) {
            wait_async0();
            __syncthreads();
            const int cur = ks & 1;
            if (ks + 1 < 32)
                async_copy_b128((unsigned)(L_WSL + (1 - cur)*4096 + t*16),
                                w116 + (size_t)(ks + 1) * 64 * 32,
                                (unsigned)(t*16));

            const bf16* arow = comb + (mt*16 + m) * CSTR + ks*32 + hi*16;
            v16bf a;
            #pragma unroll
            for (int i = 0; i < 8; ++i) { a[i] = arow[i]; a[8+i] = arow[8 + i]; }

            const bf16* brow = wsl + cur*2048 + (nt*16 + m) * 32 + hi*16;
            v16bf b;
            #pragma unroll
            for (int i = 0; i < 16; ++i) b[i] = brow[i];

            acc = __builtin_amdgcn_wmma_f32_16x16x32_bf16(
                false, a, false, b, (short)0, acc, false, false);
        }
        int col = nt*16 + m;
        float b1v = b1[col];
        #pragma unroll
        for (int r = 0; r < 8; ++r)
            hls[(mt*16 + hi*8 + r) * HSTR + col] = fmaxf(acc[r] + b1v, 0.0f);
    }
    __syncthreads();

    // ---- Phase E: credit = sigmoid(h @ W2 + b2) -----------------------------
    {
        int r = t >> 3, part = t & 7;
        const float* hrow = hls + r * HSTR + part * 8;
        const float* w2   = W2 + part * 8;
        float s = 0.f;
        #pragma unroll
        for (int j = 0; j < 8; ++j) s += hrow[j] * w2[j];
        #pragma unroll
        for (int off = 1; off < 8; off <<= 1) s += __shfl_xor(s, off, 8);
        if (part == 0) {
            float z = s + b2[0];
            credit_out[rowBase + r] = 1.0f / (1.0f + __expf(-z));
        }
    }
}

// ---------------------------------------------------------------------------
extern "C" void kernel_launch(void* const* d_in, const int* in_sizes, int n_in,
                              void* d_out, int out_size, void* d_ws, size_t ws_size,
                              hipStream_t stream) {
    const float* x   = (const float*)d_in[0];
    const int*   seq = (const int*)  d_in[1];
    const float* emb = (const float*)d_in[2];
    const float* Wq  = (const float*)d_in[3];
    const float* bq  = (const float*)d_in[4];
    const float* Wk  = (const float*)d_in[5];
    const float* bk  = (const float*)d_in[6];
    const float* Wv  = (const float*)d_in[7];
    const float* bv  = (const float*)d_in[8];
    const float* W1  = (const float*)d_in[9];
    const float* b1  = (const float*)d_in[10];
    const float* W2  = (const float*)d_in[11];
    const float* b2  = (const float*)d_in[12];

    char*  ws    = (char*)d_ws;
    float* kbuf  = (float*)(ws + WS_KBUF);
    float* vbuf  = (float*)(ws + WS_VBUF);
    bf16*  wq16  = (bf16*) (ws + WS_WQ);
    bf16*  w116  = (bf16*) (ws + WS_W1);

    float* credit_out = (float*)d_out;            // 64*4096 f32
    float* attn_out   = credit_out + 64 * 4096;   // 64*4096*4 f32

    (void)hipFuncSetAttribute((const void*)credit_main,
        hipFuncAttributeMaxDynamicSharedMemorySize, SMEM_BYTES);

    conv_weights<<<(512*512 + 1024*64 + 255)/256, 256, 0, stream>>>(Wq, W1, wq16, w116);
    prep_kv<<<1, 512, 0, stream>>>(seq, emb, Wk, bk, Wv, bv, kbuf, vbuf);

    const int nBlocks = (64 * 4096) / MT;  // 8192
    credit_main<<<nBlocks, 256, SMEM_BYTES, stream>>>(
        x, wq16, bq, kbuf, vbuf, w116, b1, W2, b2, credit_out, attn_out);
}
